// BitLinear_31224412242446
// MI455X (gfx1250) — 8-bit matrix operand order repaired and checked by reading, compiled, not run on hardware
//
#include <hip/hip_runtime.h>
#include <stdint.h>

typedef int v8i __attribute__((ext_vector_type(8)));

#define GROUPS   8
#define IN_F     4096
#define OUT_F    4096
#define MTOT     8192
#define GS       (OUT_F / GROUPS)
#define QBF      128.0f
#define CLIP_EPS 1e-6f

#define BM 128
#define BN 128
#define BK 64
#define LDST 80

__global__ __launch_bounds__(256)
void bitlinear_stats(const float* __restrict__ W,
                     float* __restrict__ alpha, float* __restrict__ beta) {
    const int g   = blockIdx.x;
    const int tid = threadIdx.x;
    const float4* base = (const float4*)(W + (size_t)g * GS * IN_F);
    const int n4 = GS * IN_F / 4;
    float s = 0.f, sa = 0.f;
    for (int i = tid; i < n4; i += 256) {
        float4 v = base[i];
        s  += (v.x + v.y) + (v.z + v.w);
        sa += (fabsf(v.x) + fabsf(v.y)) + (fabsf(v.z) + fabsf(v.w));
    }
    __shared__ float sh_s[256];
    __shared__ float sh_a[256];
    sh_s[tid] = s; sh_a[tid] = sa;
    __syncthreads();
    for (int off = 128; off > 0; off >>= 1) {
        if (tid < off) { sh_s[tid] += sh_s[tid + off]; sh_a[tid] += sh_a[tid + off]; }
        __syncthreads();
    }
    if (tid == 0) {
        const float inv = 1.0f / (float)(GS * IN_F);
        alpha[g] = sh_s[0] * inv;
        beta[g]  = sh_a[0] * inv;
    }
}

__global__ __launch_bounds__(256)
void bitlinear_binarize(const float* __restrict__ W,
                        const float* __restrict__ alpha,
                        uint32_t* __restrict__ Wb) {
    const size_t i4  = (size_t)blockIdx.x * 256 + threadIdx.x;
    const int    row = (int)(i4 >> 10);
    const float  a   = alpha[row >> 9];
    const float4 w   = ((const float4*)W)[i4];
    uint32_t p;
    p  = (uint32_t)(uint8_t)((w.x > a) ? 1 : -1);
    p |= (uint32_t)(uint8_t)((w.y > a) ? 1 : -1) << 8;
    p |= (uint32_t)(uint8_t)((w.z > a) ? 1 : -1) << 16;
    p |= (uint32_t)(uint8_t)((w.w > a) ? 1 : -1) << 24;
    Wb[i4] = p;
}

__global__ __launch_bounds__(256)
void bitlinear_quantize(const float* __restrict__ x,
                        uint32_t* __restrict__ q,
                        float* __restrict__ gamma) {
    const int token = blockIdx.x;
    const int tid   = threadIdx.x;
    const float4* xr = (const float4*)(x + (size_t)token * IN_F);
    float mx = 0.f;
    for (int i = tid; i < IN_F / 4; i += 256) {
        float4 v = xr[i];
        mx = fmaxf(mx, fmaxf(fmaxf(fabsf(v.x), fabsf(v.y)),
                             fmaxf(fabsf(v.z), fabsf(v.w))));
    }
    __shared__ float sh[256];
    sh[tid] = mx;
    __syncthreads();
    for (int off = 128; off > 0; off >>= 1) {
        if (tid < off) sh[tid] = fmaxf(sh[tid], sh[tid + off]);
        __syncthreads();
    }
    const float g = fmaxf(sh[0], CLIP_EPS);
    if (tid == 0) gamma[token] = g;
    const float s  = QBF / g;
    const float lo = -QBF + CLIP_EPS, hi = QBF - CLIP_EPS;
    for (int i = tid; i < IN_F / 4; i += 256) {
        float4 v = xr[i];
        float f[4] = { v.x, v.y, v.z, v.w };
        uint32_t p = 0;
        #pragma unroll
        for (int j = 0; j < 4; ++j) {
            float t = fminf(fmaxf(f[j] * s, lo), hi);
            int r = (int)rintf(t);
            r = (r > 127) ? 127 : r;
            p |= ((uint32_t)(uint8_t)(int8_t)r) << (8 * j);
        }
        q[(size_t)token * (IN_F / 4) + i] = p;
    }
}

__global__ __launch_bounds__(256)
void bitlinear_wmma_gemm(const uint8_t* __restrict__ q,
                         const uint8_t* __restrict__ Wb,
                         const float* __restrict__ gamma,
                         const float* __restrict__ beta,
                         const float* __restrict__ bias,
                         float* __restrict__ out)
{
    __shared__ __align__(16) uint8_t lA[BM * LDST];
    __shared__ __align__(16) uint8_t lB[BN * LDST];

    const int tid  = threadIdx.x;
    const int lane = tid & 31;
    const int wave = tid >> 5;
    const int wm   = wave >> 1;
    const int wn   = wave & 1;
    const int l16  = lane & 15;
    const int h    = lane >> 4;

    const int N0 = blockIdx.x * BN;
    const int M0 = blockIdx.y * BM;

    v8i c[2][4];
    #pragma unroll
    for (int mi = 0; mi < 2; ++mi)
        #pragma unroll
        for (int ni = 0; ni < 4; ++ni)
            #pragma unroll
            for (int e = 0; e < 8; ++e) c[mi][ni][e] = 0;

    const int ldrow = tid >> 2;
    const int ldk   = (tid & 3) * 16;

    for (int k0 = 0; k0 < IN_F; k0 += BK) {
        #pragma unroll
        for (int r = 0; r < 2; ++r) {
            const int row = ldrow + r * 64;
            *(uint4*)(lA + row * LDST + ldk) =
                *(const uint4*)(q  + (size_t)(M0 + row) * IN_F + k0 + ldk);
            *(uint4*)(lB + row * LDST + ldk) =
                *(const uint4*)(Wb + (size_t)(N0 + row) * IN_F + k0 + ldk);
        }
        if (k0 + BK < IN_F) {
            __builtin_prefetch(q  + (size_t)(M0 + ldrow) * IN_F + k0 + BK + ldk, 0, 1);
            __builtin_prefetch(Wb + (size_t)(N0 + ldrow) * IN_F + k0 + BK + ldk, 0, 1);
        }
        __syncthreads();

        v8i a[2];
        #pragma unroll
        for (int mi = 0; mi < 2; ++mi) {
            const int row = wm * 32 + mi * 16 + l16;
            const int* p = (const int*)(lA + row * LDST + 16 * h);
            v8i t;
            t[0] = p[0];  t[1] = p[1];
            t[2] = p[2];  t[3] = p[3];
            t[4] = p[8];  t[5] = p[9];
            t[6] = p[10]; t[7] = p[11];
            a[mi] = t;
        }
        v8i b[4];
        #pragma unroll
        for (int ni = 0; ni < 4; ++ni) {
            const int col = wn * 64 + ni * 16 + l16;
            const int* p = (const int*)(lB + col * LDST + 16 * h);
            v8i t;
            t[0] = p[0]; t[1] = p[1]; t[2] = p[2];  t[3] = p[3];
            t[4] = p[8]; t[5] = p[9]; t[6] = p[10]; t[7] = p[11];
            b[ni] = t;
        }

        #pragma unroll
        for (int mi = 0; mi < 2; ++mi)
            #pragma unroll
            for (int ni = 0; ni < 4; ++ni)
                c[mi][ni] = __builtin_amdgcn_wmma_i32_16x16x64_iu8(
                    true, a[mi], true, b[ni], c[mi][ni], false, false);

        __syncthreads();
    }

    #pragma unroll
    for (int mi = 0; mi < 2; ++mi) {
        const int mbase = M0 + wm * 32 + mi * 16 + 8 * h;
        float gsc[8];
        #pragma unroll
        for (int r = 0; r < 8; ++r) gsc[r] = gamma[mbase + r] * (1.0f / 128.0f);
        #pragma unroll
        for (int ni = 0; ni < 4; ++ni) {
            const int   n  = N0 + wn * 64 + ni * 16 + l16;
            const float bn = beta[n >> 9];
            const float bi = bias[n];
            #pragma unroll
            for (int r = 0; r < 8; ++r) {
                out[(size_t)(mbase + r) * OUT_F + n] =
                    (float)c[mi][ni][r] * bn * gsc[r] + bi;
            }
        }
    }
}

extern "C" void kernel_launch(void* const* d_in, const int* in_sizes, int n_in,
                              void* d_out, int out_size, void* d_ws, size_t ws_size,
                              hipStream_t stream) {
    (void)in_sizes; (void)n_in; (void)out_size; (void)ws_size;

    const float* x    = (const float*)d_in[0];
    const float* W    = (const float*)d_in[1];
    const float* bias = (const float*)d_in[2];
    float* out = (float*)d_out;

    uint8_t* ws = (uint8_t*)d_ws;
    float*   alpha = (float*)(ws + 0);
    float*   betap = (float*)(ws + 256);
    float*   gamma = (float*)(ws + 512);
    uint8_t* Wb    = ws + 65536;
    uint8_t* qq    = ws + 65536 + (size_t)OUT_F * IN_F;

    bitlinear_stats   <<<GROUPS, 256, 0, stream>>>(W, alpha, betap);
    bitlinear_binarize<<<(OUT_F * IN_F / 4) / 256, 256, 0, stream>>>(W, alpha, (uint32_t*)Wb);
    bitlinear_quantize<<<MTOT, 256, 0, stream>>>(x, (uint32_t*)qq, gamma);
    bitlinear_wmma_gemm<<<dim3(OUT_F / BN, MTOT / BM), 256, 0, stream>>>(
        qq, Wb, gamma, betap, bias, out);
}
